// ModifiedAttentionBlock_16896401343013
// MI455X (gfx1250) — compile-verified, matrix operand order checked by reading, not run on hardware
//
#include <hip/hip_runtime.h>
#include <hip/hip_bf16.h>

// ---------------- problem constants ----------------
#define BB   8
#define SS   2048
#define DD   1024
#define INV_TEMP 0.03125f   // 1/sqrt(1024)

typedef __attribute__((ext_vector_type(16))) __bf16 v16bf;
typedef __attribute__((ext_vector_type(8)))  float  v8f;
typedef __attribute__((ext_vector_type(4)))  unsigned int v4u;
typedef __attribute__((ext_vector_type(8)))  int          v8i_t;
typedef __attribute__((ext_vector_type(4)))  int          v4i_t;

// ---------------------------------------------------------------------------
// Fragment loader: LDS tile is 128 rows x 32 cols of bf16 (row-major, 64B rows).
// A-fragment (16x32) and B-fragment-transposed (N-major) share the same
// per-lane layout on CDNA5:
//   lanes 0-15 : row = lane,    VGPR 0-3 -> K 0..7,  VGPR 4-7 -> K 16..23
//   lanes 16-31: row = lane-16, VGPR 0-3 -> K 8..15, VGPR 4-7 -> K 24..31
// i.e. two contiguous 16-byte LDS chunks per lane (ds_load_b128 x2).
// ---------------------------------------------------------------------------
__device__ __forceinline__ v16bf ldfrag(const __bf16* sm, int rowBase) {
    const int lane = threadIdx.x & 31;
    const int row  = rowBase + (lane & 15);
    const int half = (lane >> 4) & 1;
    const uint4* p = reinterpret_cast<const uint4*>(sm + row * 32);
    union { v16bf v; uint4 q[2]; } f;
    f.q[0] = p[half];        // K pairs [half*8 ..]
    f.q[1] = p[2 + half];    // K pairs [16+half*8 ..]
    return f.v;
}

// 8 WMMAs updating a wave's 64x32 output region.
__device__ __forceinline__ void wmma_update(const __bf16* As, const __bf16* Bs,
                                            int waveM, int waveN, v8f acc[4][2]) {
    v16bf a[4], b[2];
#pragma unroll
    for (int tm = 0; tm < 4; ++tm) a[tm] = ldfrag(As, waveM + tm * 16);
#pragma unroll
    for (int tn = 0; tn < 2; ++tn) b[tn] = ldfrag(Bs, waveN + tn * 16);
#pragma unroll
    for (int tm = 0; tm < 4; ++tm)
#pragma unroll
        for (int tn = 0; tn < 2; ++tn)
            acc[tm][tn] = __builtin_amdgcn_wmma_f32_16x16x32_bf16(
                false, a[tm], false, b[tn], (short)0, acc[tm][tn], false, false);
}

// ---------------------------------------------------------------------------
// TDM: one descriptor = one 128-row x 32-col bf16 tile, global -> LDS.
// D# packing per CDNA5 ISA ch.8 (group0: count/lds/global/type, group1:
// data_size, tensor dims (set large: never OOB here), tile dims, row stride).
// This toolchain exposes the 6-arg builtin: (g0, g1, g2, g3, g_extra, cpol).
// ---------------------------------------------------------------------------
__device__ __forceinline__ void tdm_load_tile(unsigned int ldsAddr, unsigned long long gAddr,
                                              unsigned int rowStrideElems) {
    v4u   g0;
    v8i_t g1;
    v4i_t g2 = (v4i_t)0;
    v4i_t g3 = (v4i_t)0;
    v8i_t g4 = (v8i_t)0;

    g0[0] = 1u;                                        // count=1, user descriptor
    g0[1] = ldsAddr;                                   // LDS byte address
    g0[2] = (unsigned int)gAddr;                       // global_addr[31:0]
    g0[3] = (unsigned int)((gAddr >> 32) & 0x1FFFFFFull) | (2u << 30);  // [56:32], type=2

    const unsigned int dim0 = 1u << 20;                // tensor dims: huge (no OOB)
    const unsigned int dim1 = 1u << 20;
    g1[0] = (int)(1u << 16);                           // data_size=1 (2 bytes/elem)
    g1[1] = (int)((dim0 & 0xFFFFu) << 16);             // tensor_dim0[15:0]
    g1[2] = (int)((dim0 >> 16) | ((dim1 & 0xFFFFu) << 16));
    g1[3] = (int)((dim1 >> 16) | (32u << 16));         // tile_dim0 = 32 (K)
    g1[4] = (int)(128u);                               // tile_dim1 = 128 rows, tile_dim2=0
    g1[5] = (int)rowStrideElems;                       // tensor_dim0_stride[31:0]
    g1[6] = 0;
    g1[7] = 0;

    __builtin_amdgcn_tensor_load_to_lds(g0, g1, g2, g3, g4, 0);
}

// ---------------- LDS staging helpers (tile is 128x32) ----------------
__device__ __forceinline__ void stage_f32(__bf16* dst, const float* src, int srcPitch) {
    for (int i = threadIdx.x; i < 128 * 8; i += 256) {      // i indexes float4
        const int row = i >> 3, c4 = (i & 7) << 2;
        const float4 f = *reinterpret_cast<const float4*>(src + (size_t)row * srcPitch + c4);
        __bf16* d = dst + row * 32 + c4;
        d[0] = (__bf16)f.x; d[1] = (__bf16)f.y; d[2] = (__bf16)f.z; d[3] = (__bf16)f.w;
    }
}
// dst[n][k] = src[k][n]  (for staging V as the N-major B operand)
__device__ __forceinline__ void stage_bf16_T(__bf16* dst, const __bf16* src, int srcPitch) {
    for (int i = threadIdx.x; i < 4096; i += 256) {
        const int n = i & 127, k = i >> 7;
        dst[n * 32 + k] = src[(size_t)k * srcPitch + n];
    }
}

// ---------------------------------------------------------------------------
// Kernel 1: Qg/Kg/V projections.  z = b*3 + which (0=Q,1=K,2=V).
// C[m,n] = sum_d X[b,m,d] * W[n,d] + bias[n]    (* gates[b,m] for Q,K)
// ---------------------------------------------------------------------------
__global__ void __launch_bounds__(256)
qkv_proj_kernel(const float* __restrict__ X, const float* __restrict__ gates,
                const float* __restrict__ Wq, const float* __restrict__ bq,
                const float* __restrict__ Wk, const float* __restrict__ bk,
                const float* __restrict__ Wv, const float* __restrict__ bv,
                __bf16* __restrict__ Qg, __bf16* __restrict__ Kg, __bf16* __restrict__ V) {
    __shared__ __align__(16) __bf16 As[128 * 32];
    __shared__ __align__(16) __bf16 Bs[128 * 32];

    const int b     = blockIdx.z / 3;
    const int which = blockIdx.z % 3;
    const float*  W    = (which == 0) ? Wq : (which == 1) ? Wk : Wv;
    const float*  bias = (which == 0) ? bq : (which == 1) ? bk : bv;
    __bf16*       out  = (which == 0) ? Qg : (which == 1) ? Kg : V;
    const bool gated = (which < 2);

    const int m0 = blockIdx.x * 128;              // S dim
    const int n0 = blockIdx.y * 128;              // E dim
    const int wid = threadIdx.x >> 5;
    const int waveM = (wid >> 2) * 64;
    const int waveN = (wid & 3) * 32;

    v8f acc[4][2];
#pragma unroll
    for (int tm = 0; tm < 4; ++tm)
#pragma unroll
        for (int tn = 0; tn < 2; ++tn) acc[tm][tn] = (v8f)0.0f;

    const float* Ab = X + (size_t)b * SS * DD + (size_t)m0 * DD;
    const float* Bb = W + (size_t)n0 * DD;
    for (int k0 = 0; k0 < DD; k0 += 32) {
        __syncthreads();
        stage_f32(As, Ab + k0, DD);
        stage_f32(Bs, Bb + k0, DD);
        __syncthreads();
        wmma_update(As, Bs, waveM, waveN, acc);
    }

    const int lane = threadIdx.x & 31;
    const int half = lane >> 4, nl = lane & 15;
#pragma unroll
    for (int tm = 0; tm < 4; ++tm)
#pragma unroll
        for (int tn = 0; tn < 2; ++tn)
#pragma unroll
            for (int r = 0; r < 8; ++r) {
                const int m = m0 + waveM + tm * 16 + half * 8 + r;
                const int n = n0 + waveN + tn * 16 + nl;
                float v = acc[tm][tn][r] + bias[n];
                if (gated) v *= gates[(size_t)b * SS + m];
                out[(size_t)b * SS * DD + (size_t)m * DD + n] = (__bf16)v;
            }
}

// ---------------------------------------------------------------------------
// Kernel 2: scores[b,i,j] = (Qg[b,i,:] . Kg[b,j,:]) / 32   (f32, into d_out)
// Double-buffered: TDM (tensor_load_to_lds) streams bf16 tiles into LDS
// while all 8 waves run WMMA on the other buffer.  Wave 0 issues descriptors
// and gates completion with s_wait_tensorcnt; barriers publish LDS to the WG.
// ---------------------------------------------------------------------------
__global__ void __launch_bounds__(256)
scores_kernel(const __bf16* __restrict__ Qg, const __bf16* __restrict__ Kg,
              float* __restrict__ scores) {
    __shared__ __align__(16) __bf16 As[2][128 * 32];
    __shared__ __align__(16) __bf16 Bs[2][128 * 32];

    const int b  = blockIdx.z;
    const int m0 = blockIdx.x * 128;
    const int n0 = blockIdx.y * 128;
    const int wid = threadIdx.x >> 5;
    const int waveM = (wid >> 2) * 64;
    const int waveN = (wid & 3) * 32;
    const bool issuer = (wid == 0);

    v8f acc[4][2];
#pragma unroll
    for (int tm = 0; tm < 4; ++tm)
#pragma unroll
        for (int tn = 0; tn < 2; ++tn) acc[tm][tn] = (v8f)0.0f;

    const unsigned long long aBase =
        (unsigned long long)(uintptr_t)(Qg + (size_t)b * SS * DD + (size_t)m0 * DD);
    const unsigned long long bBase =
        (unsigned long long)(uintptr_t)(Kg + (size_t)b * SS * DD + (size_t)n0 * DD);
    const unsigned int ldsA[2] = { (unsigned int)(uintptr_t)&As[0][0],
                                   (unsigned int)(uintptr_t)&As[1][0] };
    const unsigned int ldsB[2] = { (unsigned int)(uintptr_t)&Bs[0][0],
                                   (unsigned int)(uintptr_t)&Bs[1][0] };

    const int NI = DD / 32;                       // 32 K-steps
    if (issuer) {                                 // prologue: tiles for step 0
        tdm_load_tile(ldsA[0], aBase, DD);
        tdm_load_tile(ldsB[0], bBase, DD);
    }
    for (int i = 0; i < NI; ++i) {
        const int cur = i & 1;
        if (issuer) {
            if (i + 1 < NI) {                     // prefetch next step's tiles
                const unsigned long long off = (unsigned long long)(i + 1) * 64ull; // 32 elems * 2B
                tdm_load_tile(ldsA[cur ^ 1], aBase + off, DD);
                tdm_load_tile(ldsB[cur ^ 1], bBase + off, DD);
                __builtin_amdgcn_s_wait_tensorcnt(2);   // current tiles landed
            } else {
                __builtin_amdgcn_s_wait_tensorcnt(0);
            }
        }
        __syncthreads();                          // publish LDS to all waves
        wmma_update(As[cur], Bs[cur], waveM, waveN, acc);
        __syncthreads();                          // done reading before overwrite
    }

    const int lane = threadIdx.x & 31;
    const int half = lane >> 4, nl = lane & 15;
#pragma unroll
    for (int tm = 0; tm < 4; ++tm)
#pragma unroll
        for (int tn = 0; tn < 2; ++tn)
#pragma unroll
            for (int r = 0; r < 8; ++r) {
                const int m = m0 + waveM + tm * 16 + half * 8 + r;
                const int n = n0 + waveN + tn * 16 + nl;
                scores[(size_t)b * SS * SS + (size_t)m * SS + n] = acc[tm][tn][r] * INV_TEMP;
            }
}

// ---------------------------------------------------------------------------
// Kernel 3: in-place row softmax over S elements; one block per (b,row).
// ---------------------------------------------------------------------------
__global__ void __launch_bounds__(256)
softmax_kernel(float* __restrict__ attn) {
    __shared__ float red[256];
    float* row = attn + (size_t)blockIdx.x * SS;
    const int tid = threadIdx.x;

    float mx = -3.402823466e38f;
    for (int i = tid; i < SS; i += 256) mx = fmaxf(mx, row[i]);
    red[tid] = mx; __syncthreads();
    for (int s = 128; s > 0; s >>= 1) {
        if (tid < s) red[tid] = fmaxf(red[tid], red[tid + s]);
        __syncthreads();
    }
    mx = red[0]; __syncthreads();

    float sum = 0.0f;
    for (int i = tid; i < SS; i += 256) {
        const float e = __expf(row[i] - mx);
        row[i] = e;
        sum += e;
    }
    red[tid] = sum; __syncthreads();
    for (int s = 128; s > 0; s >>= 1) {
        if (tid < s) red[tid] += red[tid + s];
        __syncthreads();
    }
    const float inv = 1.0f / red[0];
    __syncthreads();
    for (int i = tid; i < SS; i += 256) row[i] *= inv;
}

// ---------------------------------------------------------------------------
// Kernel 4: out[b,m,n] = sum_k attn[b,m,k] * V[b,k,n]
// (A needs f32->bf16 convert, B needs transpose -> keep explicit staging.)
// ---------------------------------------------------------------------------
__global__ void __launch_bounds__(256)
attnv_kernel(const float* __restrict__ attn, const __bf16* __restrict__ V,
             float* __restrict__ out) {
    __shared__ __align__(16) __bf16 As[128 * 32];
    __shared__ __align__(16) __bf16 Bs[128 * 32];

    const int b  = blockIdx.z;
    const int m0 = blockIdx.x * 128;              // S dim
    const int n0 = blockIdx.y * 128;              // E dim
    const int wid = threadIdx.x >> 5;
    const int waveM = (wid >> 2) * 64;
    const int waveN = (wid & 3) * 32;

    v8f acc[4][2];
#pragma unroll
    for (int tm = 0; tm < 4; ++tm)
#pragma unroll
        for (int tn = 0; tn < 2; ++tn) acc[tm][tn] = (v8f)0.0f;

    const float*  Ab = attn + (size_t)b * SS * SS + (size_t)m0 * SS;
    const __bf16* Vb = V + (size_t)b * SS * DD;
    for (int k0 = 0; k0 < SS; k0 += 32) {
        __syncthreads();
        stage_f32(As, Ab + k0, SS);
        stage_bf16_T(Bs, Vb + (size_t)k0 * DD + n0, DD);
        __syncthreads();
        wmma_update(As, Bs, waveM, waveN, acc);
    }

    const int lane = threadIdx.x & 31;
    const int half = lane >> 4, nl = lane & 15;
#pragma unroll
    for (int tm = 0; tm < 4; ++tm)
#pragma unroll
        for (int tn = 0; tn < 2; ++tn)
#pragma unroll
            for (int r = 0; r < 8; ++r) {
                const int m = m0 + waveM + tm * 16 + half * 8 + r;
                const int n = n0 + waveN + tn * 16 + nl;
                out[(size_t)b * SS * DD + (size_t)m * DD + n] = acc[tm][tn][r];
            }
}

// ---------------------------------------------------------------------------
extern "C" void kernel_launch(void* const* d_in, const int* in_sizes, int n_in,
                              void* d_out, int out_size, void* d_ws, size_t ws_size,
                              hipStream_t stream) {
    const float* X     = (const float*)d_in[0];
    const float* gates = (const float*)d_in[1];
    const float* Wq    = (const float*)d_in[2];
    const float* bq    = (const float*)d_in[3];
    const float* Wk    = (const float*)d_in[4];
    const float* bk    = (const float*)d_in[5];
    const float* Wv    = (const float*)d_in[6];
    const float* bv    = (const float*)d_in[7];

    float* out  = (float*)d_out;                               // (B,S,D) f32
    float* attn = out + (size_t)BB * SS * DD;                  // (B,S,S) f32

    const size_t NE = (size_t)BB * SS * DD;                    // elems per bf16 tensor
    __bf16* Qg = (__bf16*)d_ws;
    __bf16* Kg = Qg + NE;
    __bf16* Vt = Kg + NE;

    dim3 blk(256);
    qkv_proj_kernel<<<dim3(SS / 128, DD / 128, BB * 3), blk, 0, stream>>>(
        X, gates, Wq, bq, Wk, bk, Wv, bv, Qg, Kg, Vt);
    scores_kernel<<<dim3(SS / 128, SS / 128, BB), blk, 0, stream>>>(Qg, Kg, attn);
    softmax_kernel<<<dim3(BB * SS), blk, 0, stream>>>(attn);
    attnv_kernel<<<dim3(SS / 128, DD / 128, BB), blk, 0, stream>>>(attn, Vt, out);
}
